// CumulativeLinearMultiheadAttentionKV_21706764714123
// MI455X (gfx1250) — matrix operand order repaired and checked by reading, not compiled, not run on hardware
//
#include <hip/hip_runtime.h>
#include <stdint.h>

// ---------------------------------------------------------------------------
// CDNA5 / gfx1250 wave32 WMMA implementation of CumulativeLinearMultiheadAttentionKV
// T=1024, B=2, E=1024, H=16, D=64. fp32 I/O at the edges, f16 intermediates +
// pre-converted f16 weights, f16 WMMA (16x16x32) with fp32 accumulation,
// double-buffered async global->LDS tile DMA (ASYNCcnt), transposed v/S
// layouts so all hot LDS fragment reads are contiguous.
// ---------------------------------------------------------------------------

typedef __attribute__((ext_vector_type(16))) _Float16 v16h;
typedef __attribute__((ext_vector_type(8)))  float    v8f;

__device__ __forceinline__ v8f wmma16x16x32(v16h a, v16h b, v8f c) {
  return __builtin_amdgcn_wmma_f32_16x16x32_f16(false, a, false, b, (short)0, c,
                                                false, false);
}

// Async DMA of 16 bytes global -> LDS (ASYNCcnt path, CDNA5).
// lds_off is a byte offset from the workgroup's LDS base (single static block).
__device__ __forceinline__ void async_copy_b128(uint32_t lds_off, const void* gptr) {
  asm volatile("global_load_async_to_lds_b128 %0, %1, off"
               :: "v"(lds_off), "v"(gptr) : "memory");
}
__device__ __forceinline__ void wait_asynccnt0() {
  asm volatile("s_wait_asynccnt 0" ::: "memory");
}

// ---- fragment builders (CDNA5 ISA 7.12.2 layouts, wave32) -------------------
// A 16x32 (MxK), row-major source: lane m = lane&15;
// lanes 0-15 hold K 0..7 & 16..23, lanes 16-31 hold K 8..15 & 24..31.
template <typename T>
__device__ __forceinline__ v16h frag_a_rm(const T* base, int pitch, int lane) {
  const int m  = lane & 15;
  const int k0 = (lane & 16) ? 8 : 0;
  const T* p = base + m * pitch + k0;
  v16h f;
#pragma unroll
  for (int i = 0; i < 8; ++i) f[i] = (_Float16)p[i];
#pragma unroll
  for (int i = 0; i < 8; ++i) f[8 + i] = (_Float16)p[16 + i];
  return f;
}

// A = stored^T : logical A[m,k] = stored[k][m] (stored is K rows x M cols)
template <typename T>
__device__ __forceinline__ v16h frag_a_tr(const T* base, int pitch, int lane) {
  const int m  = lane & 15;
  const int k0 = (lane & 16) ? 8 : 0;
  const T* p = base + k0 * pitch + m;
  v16h f;
#pragma unroll
  for (int i = 0; i < 8; ++i) f[i] = (_Float16)p[i * pitch];
#pragma unroll
  for (int i = 0; i < 8; ++i) f[8 + i] = (_Float16)p[(16 + i) * pitch];
  return f;
}

// B 32x16 (KxN) where column n of B is stored as row n (N x K row-major src):
// lane n = lane&15; lanes 0-15 hold K 0..7 & 16..23, lanes 16-31 hold K 8..15 & 24..31.
template <typename T>
__device__ __forceinline__ v16h frag_b_nk(const T* base, int pitch, int lane) {
  const int n  = lane & 15;
  const int k0 = (lane & 16) ? 8 : 0;
  const T* p = base + n * pitch + k0;
  v16h f;
#pragma unroll
  for (int i = 0; i < 8; ++i) f[i] = (_Float16)p[i];
#pragma unroll
  for (int i = 0; i < 8; ++i) f[8 + i] = (_Float16)p[16 + i];
  return f;
}

// ---------------------------------------------------------------------------
// One-shot fp32 -> f16 weight conversion (weights are reused by 512 blocks).
// ---------------------------------------------------------------------------
__global__ __launch_bounds__(256) void cvt_f32_f16(const float* __restrict__ s,
                                                   _Float16* __restrict__ d, int n4) {
  const int i = blockIdx.x * 256 + threadIdx.x;
  if (i < n4) {
    const float4 x = ((const float4*)s)[i];
    _Float16* o = d + (size_t)i * 4;
    o[0] = (_Float16)x.x; o[1] = (_Float16)x.y;
    o[2] = (_Float16)x.z; o[3] = (_Float16)x.w;
  }
}

// ---- GEMM LDS layout: double-buffered A and W tiles + transpose staging -----
static constexpr int GPIT = 72;  // f16 pitch for 64-wide K tiles (36m%64 distinct)
static constexpr uint32_t GA0_OFF = 0;
static constexpr uint32_t GA1_OFF = GA0_OFF + 64 * GPIT * 2;   // 9216
static constexpr uint32_t GW0_OFF = GA1_OFF + 64 * GPIT * 2;   // 18432
static constexpr uint32_t GW1_OFF = GW0_OFF + 64 * GPIT * 2;   // 27648
static constexpr uint32_t GT_OFF  = GW1_OFF + 64 * GPIT * 2;   // 36864
static constexpr uint32_t GSMEM_BYTES = GT_OFF + 64 * 64 * 2;  // 45056

// async-DMA one 64x64 f16 tile (rows r0.., K-cols kb..kb+63) into LDS buffer
__device__ __forceinline__ void stage_async_tile(uint32_t buf_off, const _Float16* S,
                                                 int r0, int K, int kb, int tid) {
  for (int i = tid; i < 64 * 8; i += 128) {
    const int r = i >> 3, c8 = (i & 7) << 3;
    async_copy_b128(buf_off + (uint32_t)(r * GPIT + c8) * 2,
                    S + (size_t)(r0 + r) * K + kb + c8);
  }
}
// fp32 A pipeline: global -> registers (issued early) ...
__device__ __forceinline__ void load_a_regs(float4* regs, const float* A,
                                            int m0, int K, int kb, int tid) {
#pragma unroll
  for (int i = 0; i < 8; ++i) {
    const int idx = tid + i * 128;
    const int r = idx >> 4, c4 = (idx & 15) << 2;
    regs[i] = *(const float4*)(A + (size_t)(m0 + r) * K + kb + c4);
  }
}
// ... then registers -> LDS (cvt) after the compute phase
__device__ __forceinline__ void store_a_regs(_Float16* lA, const float4* regs, int tid) {
#pragma unroll
  for (int i = 0; i < 8; ++i) {
    const int idx = tid + i * 128;
    const int r = idx >> 4, c4 = (idx & 15) << 2;
    _Float16* d = lA + r * GPIT + c4;
    d[0] = (_Float16)regs[i].x; d[1] = (_Float16)regs[i].y;
    d[2] = (_Float16)regs[i].z; d[3] = (_Float16)regs[i].w;
  }
}

__device__ __forceinline__ void store_out(float* C, size_t idx, float v) { C[idx] = v; }
__device__ __forceinline__ void store_out(_Float16* C, size_t idx, float v) { C[idx] = (_Float16)v; }

// ---------------------------------------------------------------------------
// Tiled GEMM: C[M,N] = alpha * (A[M,K] @ W[N,K]^T + bias[N]) ; W is f16.
// 128 threads = 4 waves; 64x64 C tile; K step 64; double-buffered staging.
// TSTORE: write C transposed per head as (b, n, t) with t = row>>1, b = row&1
// (produces v^T in (B,H,D,T) layout with coalesced 64B runs).
// ---------------------------------------------------------------------------
template <typename AT, typename OT, bool TSTORE>
__global__ __launch_bounds__(128) void wmma_gemm_xWt(
    const AT* __restrict__ A, const _Float16* __restrict__ W,
    const float* __restrict__ bias, OT* __restrict__ C,
    int M, int N, int K, float alpha) {
  constexpr bool AF16 = (sizeof(AT) == 2);
  __shared__ __align__(16) unsigned char gsmem[GSMEM_BYTES];

  const int tid  = threadIdx.x;
  const int lane = tid & 31;
  const int wave = tid >> 5;
  const int n0 = blockIdx.x * 64;
  const int m0 = blockIdx.y * 64;
  const uint32_t aoff[2] = {GA0_OFF, GA1_OFF};
  const uint32_t woff[2] = {GW0_OFF, GW1_OFF};

  v8f acc[4] = {};
  float4 areg[8];

  // prologue: stage tile kb=0 into buffer 0
  if constexpr (AF16) {
    stage_async_tile(GA0_OFF, (const _Float16*)A, m0, K, 0, tid);
  } else {
    load_a_regs(areg, (const float*)A, m0, K, 0, tid);
    store_a_regs((_Float16*)(gsmem + GA0_OFF), areg, tid);
  }
  stage_async_tile(GW0_OFF, W, n0, K, 0, tid);
  wait_asynccnt0();
  __syncthreads();

  int p = 0;
  for (int kb = 0; kb < K; kb += 64) {
    const bool has_next = (kb + 64) < K;
    if (has_next) {  // issue next tile's DMA / global loads before computing
      if constexpr (AF16)
        stage_async_tile(aoff[p ^ 1], (const _Float16*)A, m0, K, kb + 64, tid);
      else
        load_a_regs(areg, (const float*)A, m0, K, kb + 64, tid);
      stage_async_tile(woff[p ^ 1], W, n0, K, kb + 64, tid);
    }

    const _Float16* lA = (const _Float16*)(gsmem + aoff[p]);
    const _Float16* lW = (const _Float16*)(gsmem + woff[p]);
    const v16h af0 = frag_a_rm(lA + wave * 16 * GPIT, GPIT, lane);
    const v16h af1 = frag_a_rm(lA + wave * 16 * GPIT + 32, GPIT, lane);
#pragma unroll
    for (int nt = 0; nt < 4; ++nt) {
      const v16h bf0 = frag_b_nk(lW + nt * 16 * GPIT, GPIT, lane);
      const v16h bf1 = frag_b_nk(lW + nt * 16 * GPIT + 32, GPIT, lane);
      acc[nt] = wmma16x16x32(af0, bf0, acc[nt]);
      acc[nt] = wmma16x16x32(af1, bf1, acc[nt]);
    }

    if (has_next) {
      if constexpr (!AF16)
        store_a_regs((_Float16*)(gsmem + aoff[p ^ 1]), areg, tid);
      wait_asynccnt0();
    }
    __syncthreads();
    p ^= 1;
  }

  // epilogue: C/D layout -> row = r + 8*(lane>=16), col = lane&15 per 16x16 tile
  const int cn   = lane & 15;
  const int rsel = (lane >> 4) << 3;
  if constexpr (TSTORE) {
    // de-interleave (t,b) and transpose through LDS; rows of v^T are coalesced
    _Float16* lT = (_Float16*)(gsmem + GT_OFF);  // [n][b][t'] : 64 x 2 x 32
    __syncthreads();
#pragma unroll
    for (int nt = 0; nt < 4; ++nt) {
      const float bv = bias[n0 + nt * 16 + cn];
#pragma unroll
      for (int r = 0; r < 8; ++r) {
        const int m = wave * 16 + rsel + r;  // global row m0+m = t*2+b
        lT[(nt * 16 + cn) * 64 + (m & 1) * 32 + (m >> 1)] =
            (_Float16)(alpha * (acc[nt][r] + bv));
      }
    }
    __syncthreads();
    const int n = tid >> 1, bs = tid & 1;
    const int Tt = M >> 1;  // sequence length
    const float4* s = (const float4*)(lT + n * 64 + bs * 32);
    float4* d = (float4*)(C + ((size_t)bs * N + n0 + n) * Tt + (m0 >> 1));
#pragma unroll
    for (int j = 0; j < 4; ++j) d[j] = s[j];
  } else {
#pragma unroll
    for (int nt = 0; nt < 4; ++nt) {
      const int col = n0 + nt * 16 + cn;
      const float bv = bias[col];
#pragma unroll
      for (int r = 0; r < 8; ++r) {
        const int row = m0 + wave * 16 + rsel + r;
        store_out(C, (size_t)row * N + col, alpha * (acc[nt][r] + bv));
      }
    }
  }
}

// ---------------------------------------------------------------------------
// Chunked causal linear attention: one block per (h, b), chunk C = 64.
// qb/kb are (T,B,E) f16 (q pre-scaled by 1/sqrt(D)); vt is v^T in (B,H,D,T) f16.
//   out_chunk = q @ S_prev + tril(q @ k^T) @ v ;  S += k^T @ v
// Chunk tiles are DMA'd global->LDS (ASYNCcnt). S accumulates in fp32 WMMA
// registers; its f16 TRANSPOSED snapshot in LDS serves the q@S B-operand as a
// contiguous row read. attn@v and k^T@v share the v^T B-fragments.
// ---------------------------------------------------------------------------
static constexpr int AP_PQ = 88;  // f16 pitch: 176B rows (16B aligned), 44m%64 distinct
static constexpr uint32_t LQ_OFF  = 0;
static constexpr uint32_t LK_OFF  = LQ_OFF + 64 * AP_PQ * 2;    // 11264
static constexpr uint32_t LV_OFF  = LK_OFF + 64 * AP_PQ * 2;    // 22528
static constexpr uint32_t LST_OFF = LV_OFF + 64 * AP_PQ * 2;    // 33792
static constexpr uint32_t SMEM_BYTES = LST_OFF + 64 * AP_PQ * 2;  // 45056

__global__ __launch_bounds__(128) void lin_attn_chunk(
    const _Float16* __restrict__ qb, const _Float16* __restrict__ kb,
    const _Float16* __restrict__ vt, _Float16* __restrict__ ob) {
  constexpr int E = 1024, Bd = 2, D = 64, CH = 64, NCH = 16, Tt = 1024;
  constexpr int PQ = AP_PQ;
  __shared__ __align__(16) unsigned char smem[SMEM_BYTES];
  _Float16* lq  = (_Float16*)(smem + LQ_OFF);   // q chunk; reused for attn probs
  _Float16* lk  = (_Float16*)(smem + LK_OFF);   // k chunk (row-major)
  _Float16* lvT = (_Float16*)(smem + LV_OFF);   // v^T chunk: rows d, cols t
  _Float16* lSt = (_Float16*)(smem + LST_OFF);  // S^T f16 snapshot: rows d, cols e

  const int tid  = threadIdx.x;
  const int lane = tid & 31;
  const int wave = tid >> 5;
  const int h = blockIdx.x;
  const int b = blockIdx.y;
  const size_t colbase = (size_t)h * D;
  const int cn   = lane & 15;
  const int rsel = (lane >> 4) << 3;

  {  // zero the S^T snapshot
    uint32_t* z = (uint32_t*)(smem + LST_OFF);
    for (int i = tid; i < 64 * PQ / 2; i += 128) z[i] = 0u;
  }

  v8f Sreg[4] = {};  // wave owns S rows e in [16*wave, 16*wave+16)

  for (int c = 0; c < NCH; ++c) {
    const int t0 = c * CH;
    __syncthreads();  // prev chunk fully consumed lq/lk/lvT; S^T snapshot done

    // async DMA of q/k (row-major) and v^T (d-major) chunk tiles into LDS
    for (int i = tid; i < CH * 8; i += 128) {
      const int r = i >> 3, j = i & 7;  // j: 16B chunk within the 128B row
      const size_t g  = ((size_t)(t0 + r) * Bd + b) * E + colbase + j * 8;
      const size_t gv = ((size_t)b * E + colbase + r) * Tt + t0 + j * 8;
      async_copy_b128(LQ_OFF + (uint32_t)(r * PQ + j * 8) * 2, qb + g);
      async_copy_b128(LK_OFF + (uint32_t)(r * PQ + j * 8) * 2, kb + g);
      async_copy_b128(LV_OFF + (uint32_t)(r * PQ + j * 8) * 2, vt + gv);
    }
    wait_asynccnt0();
    __syncthreads();

    v8f acco[4] = {};  // out rows 16*wave.., all 4 d-tiles

    // A fragments of q (rows 16w.., K = e, two 32-wide slices)
    const v16h aq0 = frag_a_rm(lq + wave * 16 * PQ, PQ, lane);
    const v16h aq1 = frag_a_rm(lq + wave * 16 * PQ + 32, PQ, lane);

    // inter-chunk: out += q @ S_prev   (B cols = rows of S^T, contiguous f16)
#pragma unroll
    for (int nt = 0; nt < 4; ++nt) {
      const v16h b0 = frag_b_nk(lSt + nt * 16 * PQ, PQ, lane);       // e 0..31
      const v16h b1 = frag_b_nk(lSt + nt * 16 * PQ + 32, PQ, lane);  // e 32..63
      acco[nt] = wmma16x16x32(aq0, b0, acco[nt]);
      acco[nt] = wmma16x16x32(aq1, b1, acco[nt]);
    }

    // intra-chunk scores: attn = q @ k^T  (B cols = rows of k, contiguous)
    v8f acca[4] = {};
#pragma unroll
    for (int nt = 0; nt < 4; ++nt) {
      const v16h b0 = frag_b_nk(lk + nt * 16 * PQ, PQ, lane);
      const v16h b1 = frag_b_nk(lk + nt * 16 * PQ + 32, PQ, lane);
      acca[nt] = wmma16x16x32(aq0, b0, acca[nt]);
      acca[nt] = wmma16x16x32(aq1, b1, acca[nt]);
    }

    __syncthreads();  // all reads of lq (as q) and lSt (old S^T) are done

    // causal mask (j <= i, inclusive) and restage probs as f16 into lq
#pragma unroll
    for (int nt = 0; nt < 4; ++nt) {
      const int j = nt * 16 + cn;
#pragma unroll
      for (int r = 0; r < 8; ++r) {
        const int i = wave * 16 + rsel + r;
        lq[i * PQ + j] = (j <= i) ? (_Float16)acca[nt][r] : (_Float16)0.0f;
      }
    }
    __syncthreads();  // probs visible

    // fused: out += attn @ v  AND  S += k^T @ v — shared v^T B-fragments
    const v16h aa0 = frag_a_rm(lq + wave * 16 * PQ, PQ, lane);
    const v16h aa1 = frag_a_rm(lq + wave * 16 * PQ + 32, PQ, lane);
    const v16h ak0 = frag_a_tr(lk + wave * 16, PQ, lane);            // j 0..31
    const v16h ak1 = frag_a_tr(lk + 32 * PQ + wave * 16, PQ, lane);  // j 32..63
#pragma unroll
    for (int nt = 0; nt < 4; ++nt) {
      const v16h bv0 = frag_b_nk(lvT + nt * 16 * PQ, PQ, lane);       // j 0..31
      const v16h bv1 = frag_b_nk(lvT + nt * 16 * PQ + 32, PQ, lane);  // j 32..63
      acco[nt] = wmma16x16x32(aa0, bv0, acco[nt]);
      acco[nt] = wmma16x16x32(aa1, bv1, acco[nt]);
      Sreg[nt] = wmma16x16x32(ak0, bv0, Sreg[nt]);
      Sreg[nt] = wmma16x16x32(ak1, bv1, Sreg[nt]);
    }

    // snapshot updated S TRANSPOSED (f16) for next chunk's q@S
#pragma unroll
    for (int nt = 0; nt < 4; ++nt)
#pragma unroll
      for (int r = 0; r < 8; ++r)
        lSt[(nt * 16 + cn) * PQ + wave * 16 + rsel + r] = (_Float16)Sreg[nt][r];

    // write out chunk in (T,B,E) f16 layout
#pragma unroll
    for (int nt = 0; nt < 4; ++nt) {
#pragma unroll
      for (int r = 0; r < 8; ++r) {
        const int t = t0 + wave * 16 + rsel + r;
        ob[((size_t)t * Bd + b) * E + colbase + nt * 16 + cn] = (_Float16)acco[nt][r];
      }
    }
  }
}

// ---------------------------------------------------------------------------
extern "C" void kernel_launch(void* const* d_in, const int* in_sizes, int n_in,
                              void* d_out, int out_size, void* d_ws, size_t ws_size,
                              hipStream_t stream) {
  (void)in_sizes; (void)n_in; (void)out_size; (void)ws_size;

  const float* query = (const float*)d_in[0];
  const float* key_  = (const float*)d_in[1];
  const float* value = (const float*)d_in[2];
  const float* ipw   = (const float*)d_in[3];  // (3E, E)
  const float* ipb   = (const float*)d_in[4];  // (3E,)
  const float* opw   = (const float*)d_in[5];  // (E, E)
  const float* opb   = (const float*)d_in[6];  // (E,)
  float* out = (float*)d_out;

  const int T = 1024, B = 2, E = 1024;
  const int M = T * B;  // 2048 rows of E
  // f16 workspace: q/k (T,B,E), v^T (B,H,D,T), attn out (T,B,E), f16 weights
  _Float16* qbuf  = (_Float16*)d_ws;
  _Float16* kbuf  = qbuf  + (size_t)M * E;
  _Float16* vtbuf = kbuf  + (size_t)M * E;
  _Float16* abuf  = vtbuf + (size_t)M * E;
  _Float16* wip16 = abuf  + (size_t)M * E;          // (3E, E) f16
  _Float16* wop16 = wip16 + (size_t)3 * E * E;      // (E, E) f16

  const dim3 blk(128);
  const dim3 gg(E / 64, M / 64);
  const float scale = 0.125f;  // 1/sqrt(D), folded into the q projection

  // one-shot weight conversion to f16
  cvt_f32_f16<<<dim3(3 * E * E / 1024), dim3(256), 0, stream>>>(ipw, wip16, 3 * E * E / 4);
  cvt_f32_f16<<<dim3(E * E / 1024),     dim3(256), 0, stream>>>(opw, wop16, E * E / 4);

  wmma_gemm_xWt<float, _Float16, false><<<gg, blk, 0, stream>>>(query, wip16, ipb, qbuf, M, E, E, scale);
  wmma_gemm_xWt<float, _Float16, false><<<gg, blk, 0, stream>>>(key_,  wip16 + (size_t)E * E, ipb + E, kbuf, M, E, E, 1.0f);
  wmma_gemm_xWt<float, _Float16, true ><<<gg, blk, 0, stream>>>(value, wip16 + (size_t)2 * E * E, ipb + 2 * E, vtbuf, M, E, E, 1.0f);

  lin_attn_chunk<<<dim3(16, 2), blk, 0, stream>>>(qbuf, kbuf, vtbuf, abuf);

  wmma_gemm_xWt<_Float16, float, false><<<gg, blk, 0, stream>>>(abuf, wop16, opb, out, M, E, E, 1.0f);
}
